// MultiHeadAttention_84705345012460
// MI455X (gfx1250) — matrix operand order repaired and checked by reading, not compiled, not run on hardware
//
#include <hip/hip_runtime.h>
#include <stdint.h>

// ---------------------------------------------------------------------------
// MI455X (gfx1250) fused QKV-projection + flash-attention, bf16 WMMA path.
//   x[2,2048,1024]f32, W[1024,3072]f32, b[3072]f32 -> out[2,2048,1024]f32
// Compute-bound (~60 GFLOP vs ~50MB traffic @ 23.3TB/s) => bf16 WMMA,
// fp32 accumulation, fp32 online softmax. Tile staging uses CDNA5
// GLOBAL_LOAD_ASYNC_TO_LDS_B128 (ASYNCcnt) when the toolchain exposes it.
// ---------------------------------------------------------------------------

#define SEQ 2048
#define EMB 1024
#define NH  16
#define DH  64
#define NBAT 2
#define MTOT (NBAT * SEQ)   // 4096 rows of x
#define N3   (3 * EMB)      // 3072 qkv columns

typedef __attribute__((ext_vector_type(16))) __bf16   v16bf;
typedef __attribute__((ext_vector_type(8)))  float    v8f;
typedef __attribute__((ext_vector_type(4)))  uint32_t u32x4;
typedef __attribute__((ext_vector_type(2)))  float    f32x2;

struct Frag32 { u32x4 lo, hi; };   // 32 bytes == one v16bf WMMA operand
static_assert(sizeof(Frag32) == 32, "frag size");
static_assert(sizeof(v16bf)  == 32, "v16bf size");

__device__ __forceinline__ v16bf as_bf(const Frag32 &f) {
  return __builtin_bit_cast(v16bf, f);
}

__device__ __forceinline__ uint16_t f2bf(float f) {   // RNE f32 -> bf16
  uint32_t u = __float_as_uint(f);
  u += 0x7FFFu + ((u >> 16) & 1u);
  return (uint16_t)(u >> 16);
}

__device__ __forceinline__ v8f wmma_bf16(v16bf a, v16bf b, v8f c) {
  // D = A(16x32 bf16) * B(32x16 bf16) + C(16x16 f32)
  return __builtin_amdgcn_wmma_f32_16x16x32_bf16(false, a, false, b,
                                                 (short)0, c, false, false);
}

// ---------------- async memory->LDS staging (CDNA5 ASYNCcnt path) ----------

#if defined(__AMDGCN__) &&                                              \
    __has_builtin(__builtin_amdgcn_global_load_async_to_lds_b128) &&    \
    __has_builtin(__builtin_amdgcn_s_wait_asynccnt)
#define HAVE_ASYNC_LDS 1
#else
#define HAVE_ASYNC_LDS 0
#endif

// Builtin signature (from clang diagnostic): arg0 is
//   '__attribute__((vector_size(16))) int __device__ *'  (AS1)
// arg1 is the LDS-side pointer (AS3), then imm offset, imm cpol.
typedef int v4i_vec __attribute__((vector_size(16)));
typedef __attribute__((address_space(1))) v4i_vec *g_v4i_p;
typedef __attribute__((address_space(3))) v4i_vec *l_v4i_p;

// Copy one 16B chunk per lane from global memory straight into LDS.
// Async path: GLOBAL_LOAD_ASYNC_TO_LDS_B128 (no VGPR bounce, ASYNCcnt).
__device__ __forceinline__ void copy16_to_lds(const uint16_t *g, uint16_t *l) {
#if HAVE_ASYNC_LDS
  // generic LDS pointer -> low 32 bits == LDS byte offset -> AS(3) pointer
  __builtin_amdgcn_global_load_async_to_lds_b128(
      (g_v4i_p)(uintptr_t)g,
      (l_v4i_p)(uint32_t)(uintptr_t)l,
      /*offset=*/0, /*cpol=*/0);
#else
  *(u32x4 *)l = *(const u32x4 *)g;
#endif
}

// Wait for this wave's async staging, then workgroup barrier.
__device__ __forceinline__ void wait_lds_staging() {
#if HAVE_ASYNC_LDS
  __builtin_amdgcn_s_wait_asynccnt(0);
#endif
  __syncthreads();
}

// --------------------------- precision converts ----------------------------

__global__ void cvt_x_kernel(const float *__restrict__ x,
                             uint32_t *__restrict__ xb, int npairs) {
  int i = blockIdx.x * blockDim.x + threadIdx.x;
  if (i >= npairs) return;
  f32x2 v = *(const f32x2 *)(x + 2 * (size_t)i);
  xb[i] = (uint32_t)f2bf(v.x) | ((uint32_t)f2bf(v.y) << 16);
}

// W[1024,3072] f32  ->  Wt[3072,1024] bf16 (so GEMM B staging is coalesced)
__global__ void cvt_wt_kernel(const float *__restrict__ W,
                              uint16_t *__restrict__ wt) {
  int i = blockIdx.x * blockDim.x + threadIdx.x;   // over N3*EMB
  int n = i >> 10;
  int k = i & 1023;
  wt[i] = f2bf(W[(size_t)k * N3 + n]);
}

// ------------------------------ QKV projection -----------------------------
// C[4096,3072] = xb @ Wt^T, tiled 128x128 per workgroup, 8 waves of 32x64.
// Q scaled by 1/sqrt(64); K row-major [bh][s][64]; V stored transposed
// [bh][64][s] so attention's PV B-operand reads contiguous k from LDS.

__launch_bounds__(256, 2)
__global__ void qkv_proj_kernel(const uint16_t *__restrict__ xb,
                                const uint16_t *__restrict__ wt,
                                const float *__restrict__ bias,
                                uint16_t *__restrict__ qb,
                                uint16_t *__restrict__ kb,
                                uint16_t *__restrict__ vtb) {
  // 32-half rows padded to 40 halves (80B): stride 20 dwords, gcd(20,64)=4
  // => 16 consecutive reading lanes hit 16 distinct banks; 80 % 16 == 0 so
  // ds_load_b128 stays naturally aligned.
  __shared__ __align__(16) uint16_t As[128 * 40];
  __shared__ __align__(16) uint16_t Bs[128 * 40];

  const int tid  = threadIdx.x;
  const int lane = tid & 31;
  const int w    = tid >> 5;          // 0..7
  const int g    = (lane >> 4) & 1;   // half-wave group
  const int l16  = lane & 15;
  const int waveM = w >> 1;           // 0..3  (rows)
  const int waveN = w & 1;            // 0..1  (cols)

  const int mBase = blockIdx.y * 128;
  const int nBase = blockIdx.x * 128;

  v8f zero = {};
  v8f acc[2][4];
#pragma unroll
  for (int mi = 0; mi < 2; ++mi)
#pragma unroll
    for (int ni = 0; ni < 4; ++ni) acc[mi][ni] = zero;

  for (int kt = 0; kt < EMB; kt += 32) {
    // ---- stage 128x32 A tile and 128x32 B tile (each 8KB, 512 b128 chunks)
    // direct memory->LDS via async b128 copies when available
#pragma unroll
    for (int it = 0; it < 2; ++it) {
      int chunk = tid + it * 256;
      int row   = chunk >> 2;
      int part  = chunk & 3;
      copy16_to_lds(&xb[(size_t)(mBase + row) * EMB + kt + part * 8],
                    &As[row * 40 + part * 8]);
      copy16_to_lds(&wt[(size_t)(nBase + row) * EMB + kt + part * 8],
                    &Bs[row * 40 + part * 8]);
    }
    wait_lds_staging();

    // ---- operand gathers per documented CDNA5 WMMA VGPR layouts
    Frag32 af[2], bf[4];
#pragma unroll
    for (int mi = 0; mi < 2; ++mi) {
      int r = waveM * 32 + mi * 16 + l16;
      af[mi].lo = *(const u32x4 *)&As[r * 40 + g * 8];        // K g*8..g*8+7
      af[mi].hi = *(const u32x4 *)&As[r * 40 + 16 + g * 8];   // K 16+g*8..
    }
#pragma unroll
    for (int ni = 0; ni < 4; ++ni) {
      int r = waveN * 64 + ni * 16 + l16;
      bf[ni].lo = *(const u32x4 *)&Bs[r * 40 + g * 8];        // K g*8..g*8+7
      bf[ni].hi = *(const u32x4 *)&Bs[r * 40 + 16 + g * 8];   // K 16+g*8..
    }
#pragma unroll
    for (int mi = 0; mi < 2; ++mi)
#pragma unroll
      for (int ni = 0; ni < 4; ++ni)
        acc[mi][ni] = wmma_bf16(as_bf(af[mi]), as_bf(bf[ni]), acc[mi][ni]);
    __syncthreads();
  }

  // ---- epilogue: bias, scatter into Q (scaled), K, Vt
  float bv[4];
#pragma unroll
  for (int ni = 0; ni < 4; ++ni)
    bv[ni] = bias[nBase + waveN * 64 + ni * 16 + l16];

#pragma unroll
  for (int mi = 0; mi < 2; ++mi) {
#pragma unroll
    for (int ni = 0; ni < 4; ++ni) {
      int n     = nBase + waveN * 64 + ni * 16 + l16;
      int which = n >> 10;          // 0=Q 1=K 2=V
      int h     = (n >> 6) & 15;
      int dh    = n & 63;
#pragma unroll
      for (int r = 0; r < 8; ++r) {
        int m = mBase + waveM * 32 + mi * 16 + r + 8 * g;  // C/D row layout
        int b = m >> 11;
        int s = m & 2047;
        int bh = b * NH + h;
        float v = acc[mi][ni][r] + bv[ni];
        if (which == 0) {
          qb[((size_t)bh * SEQ + s) * DH + dh] = f2bf(v * 0.125f); // 1/sqrt(64)
        } else if (which == 1) {
          kb[((size_t)bh * SEQ + s) * DH + dh] = f2bf(v);
        } else {
          vtb[((size_t)bh * DH + dh) * SEQ + s] = f2bf(v);         // V^T
        }
      }
    }
  }
}

// ------------------------------ flash attention ----------------------------
// grid (32 bh, 16 qTiles); 256 threads; wave w owns 16 query rows.
// Q frags live in registers for all 64 key blocks; K/V^T staged via
// async memory->LDS copies.

__launch_bounds__(256, 2)
__global__ void flash_attn_kernel(const uint16_t *__restrict__ qb,
                                  const uint16_t *__restrict__ kb,
                                  const uint16_t *__restrict__ vtb,
                                  float *__restrict__ out) {
  // 64-half K rows padded to 72 halves (144B): stride 36 dwords,
  // gcd(36,64)=4 => conflict-free; 144 % 16 == 0.
  __shared__ __align__(16) uint16_t Ks[32 * 72];       // [keypos][d]
  __shared__ __align__(16) uint16_t Vts[64 * 40];      // [d][t]   (V^T tile)
  __shared__ __align__(16) uint16_t Ps[8 * 16 * 40];   // per-wave P tile

  const int tid  = threadIdx.x;
  const int lane = tid & 31;
  const int w    = tid >> 5;
  const int g    = (lane >> 4) & 1;
  const int l16  = lane & 15;

  const int bh    = blockIdx.x;
  const int qTile = blockIdx.y;
  const uint16_t *Qh = qb  + (size_t)bh * SEQ * DH;
  const uint16_t *Kh = kb  + (size_t)bh * SEQ * DH;
  const uint16_t *Vh = vtb + (size_t)bh * DH * SEQ;

  // Q A-fragments (16x32 each, two K chunks covering d=0..63), direct global
  const int qRow = qTile * 128 + w * 16 + l16;
  Frag32 qf[2];
#pragma unroll
  for (int kk = 0; kk < 2; ++kk) {
    qf[kk].lo = *(const u32x4 *)&Qh[(size_t)qRow * DH + kk * 32 + g * 8];
    qf[kk].hi = *(const u32x4 *)&Qh[(size_t)qRow * DH + kk * 32 + 16 + g * 8];
  }

  float mrun[8], lrun[8];
  v8f zero = {};
  v8f oacc[4];
#pragma unroll
  for (int r = 0; r < 8; ++r) { mrun[r] = -1e30f; lrun[r] = 0.0f; }
#pragma unroll
  for (int d = 0; d < 4; ++d) oacc[d] = zero;

  const int kRow = tid >> 3, kPart = tid & 7;   // K staging: 32 rows x 8 chunks
  const int vRow = tid >> 2, vPart = tid & 3;   // Vt staging: 64 rows x 4 chunks

  for (int t0 = 0; t0 < SEQ; t0 += 32) {
    // ---- stage 32-key block of K and V^T (async memory->LDS)
    copy16_to_lds(&Kh[(size_t)(t0 + kRow) * DH + kPart * 8],
                  &Ks[kRow * 72 + kPart * 8]);
    copy16_to_lds(&Vh[(size_t)vRow * SEQ + t0 + vPart * 8],
                  &Vts[vRow * 40 + vPart * 8]);
    if (t0 + 32 < SEQ) {  // warm next block (global_prefetch_b8)
      __builtin_prefetch(&Kh[(size_t)(t0 + 32 + kRow) * DH + kPart * 8], 0, 0);
      __builtin_prefetch(&Vh[(size_t)vRow * SEQ + t0 + 32 + vPart * 8], 0, 0);
    }
    wait_lds_staging();

    // ---- scores: S(16x32) = Q(16x64) @ K^T, 4 WMMAs
    v8f sc0 = zero, sc1 = zero;
#pragma unroll
    for (int kk = 0; kk < 2; ++kk) {
      Frag32 b0, b1;
      int r0 = l16;        // key cols 0..15
      int r1 = 16 + l16;   // key cols 16..31
      b0.lo = *(const u32x4 *)&Ks[r0 * 72 + kk * 32 + g * 8];
      b0.hi = *(const u32x4 *)&Ks[r0 * 72 + kk * 32 + 16 + g * 8];
      b1.lo = *(const u32x4 *)&Ks[r1 * 72 + kk * 32 + g * 8];
      b1.hi = *(const u32x4 *)&Ks[r1 * 72 + kk * 32 + 16 + g * 8];
      sc0 = wmma_bf16(as_bf(qf[kk]), as_bf(b0), sc0);
      sc1 = wmma_bf16(as_bf(qf[kk]), as_bf(b1), sc1);
    }

    // ---- online softmax (fp32): 16-lane butterflies within half-wave
    uint16_t *psw = &Ps[w * 640];
#pragma unroll
    for (int r = 0; r < 8; ++r) {
      float s0 = sc0[r], s1 = sc1[r];
      float mx = fmaxf(s0, s1);
#pragma unroll
      for (int off = 1; off < 16; off <<= 1)
        mx = fmaxf(mx, __shfl_xor(mx, off, 32));
      float mnew = fmaxf(mrun[r], mx);
      float al = __expf(mrun[r] - mnew);
      float p0 = __expf(s0 - mnew);
      float p1 = __expf(s1 - mnew);
      float rs = p0 + p1;
#pragma unroll
      for (int off = 1; off < 16; off <<= 1)
        rs += __shfl_xor(rs, off, 32);
      lrun[r] = lrun[r] * al + rs;
      mrun[r] = mnew;
#pragma unroll
      for (int d = 0; d < 4; ++d) oacc[d][r] *= al;
      // D-layout element [r+8g][n] -> wave-private LDS (A-layout reload below)
      psw[(r + 8 * g) * 40 + l16]      = f2bf(p0);
      psw[(r + 8 * g) * 40 + 16 + l16] = f2bf(p1);
    }

    // ---- P as A operand (16x32), then O += P @ V (4 WMMAs over d-blocks)
    Frag32 pf;
    pf.lo = *(const u32x4 *)&psw[l16 * 40 + g * 8];
    pf.hi = *(const u32x4 *)&psw[l16 * 40 + 16 + g * 8];
#pragma unroll
    for (int d = 0; d < 4; ++d) {
      Frag32 vf;
      int rv = d * 16 + l16;                         // row of V^T = head dim
      vf.lo = *(const u32x4 *)&Vts[rv * 40 + g * 8];
      vf.hi = *(const u32x4 *)&Vts[rv * 40 + 16 + g * 8];
      oacc[d] = wmma_bf16(as_bf(pf), as_bf(vf), oacc[d]);
    }
    __syncthreads();
  }

  // ---- normalize and write fp32 output [b][s][h*64+dh]
  const int b = bh >> 4, h = bh & 15;
#pragma unroll
  for (int d = 0; d < 4; ++d) {
#pragma unroll
    for (int r = 0; r < 8; ++r) {
      int s   = qTile * 128 + w * 16 + r + 8 * g;
      int col = h * DH + d * 16 + l16;
      out[((size_t)(b * SEQ + s)) * EMB + col] = oacc[d][r] / lrun[r];
    }
  }
}

// ------------------------------- launcher ----------------------------------

extern "C" void kernel_launch(void *const *d_in, const int *in_sizes, int n_in,
                              void *d_out, int out_size, void *d_ws,
                              size_t ws_size, hipStream_t stream) {
  (void)in_sizes; (void)n_in; (void)out_size; (void)ws_size;
  const float *x    = (const float *)d_in[0];   // [2,2048,1024]
  const float *Wq   = (const float *)d_in[1];   // [1024,3072]
  const float *bias = (const float *)d_in[2];   // [3072]
  float *out = (float *)d_out;

  // workspace layout (bf16 halves), ~40 MB total
  const size_t XB  = (size_t)MTOT * EMB;        // x bf16
  const size_t WTB = (size_t)N3 * EMB;          // W^T bf16
  const size_t QKV = (size_t)NBAT * NH * SEQ * DH;
  uint16_t *xb  = (uint16_t *)d_ws;
  uint16_t *wtb = xb + XB;
  uint16_t *qb  = wtb + WTB;
  uint16_t *kb  = qb + QKV;
  uint16_t *vtb = kb + QKV;

  int npairs = (int)(XB / 2);
  cvt_x_kernel<<<npairs / 256, 256, 0, stream>>>(x, (uint32_t *)xb, npairs);
  cvt_wt_kernel<<<(int)(WTB / 256), 256, 0, stream>>>(Wq, wtb);

  dim3 gProj(N3 / 128, MTOT / 128);             // (24, 32)
  qkv_proj_kernel<<<gProj, 256, 0, stream>>>(xb, wtb, bias, qb, kb, vtb);

  dim3 gAttn(NBAT * NH, SEQ / 128);             // (32, 16)
  flash_attn_kernel<<<gAttn, 256, 0, stream>>>(qb, kb, vtb, out);
}
